// CausalSelfAttention_19473381720416
// MI455X (gfx1250) — matrix operand order repaired and checked by reading, not compiled, not run on hardware
//
#include <hip/hip_runtime.h>
#include <hip/hip_bf16.h>
#include <cstdint>
#include <cstddef>

// ---------- types ----------
typedef __bf16          v16bf  __attribute__((ext_vector_type(16)));
typedef float           v8f    __attribute__((ext_vector_type(8)));
typedef float           f32x4  __attribute__((ext_vector_type(4)));
typedef unsigned short  u16x4  __attribute__((ext_vector_type(4)));
typedef unsigned short  u16x8  __attribute__((ext_vector_type(8)));

union FragBF {
  u16x8          h[2];
  unsigned short e[16];
  v16bf          v;
};

// f32 -> bf16 via hardware convert (RNE); raw bits for ushort storage
__device__ __forceinline__ unsigned short f2bf_raw(float f) {
  __bf16 b = (__bf16)f;
  return __builtin_bit_cast(unsigned short, b);
}

// load a 16-element bf16 fragment from two 16B-aligned 8-element chunks
__device__ __forceinline__ v16bf ld_frag_pair(const unsigned short* p0,
                                              const unsigned short* p1) {
  FragBF f;
  f.h[0] = *(const u16x8*)p0;
  f.h[1] = *(const u16x8*)p1;
  return f.v;
}

__device__ __forceinline__ v8f wmma_bf(v16bf a, v16bf b, v8f c) {
  return __builtin_amdgcn_wmma_f32_16x16x32_bf16(
      /*neg_a=*/false, a, /*neg_b=*/false, b,
      /*c_mod=*/(short)0, c, /*reuse_a=*/false, /*reuse_b=*/false);
}

// ---------- CDNA5 async global->LDS copy (per-lane 16B), ASYNCcnt tracked ----
__device__ __forceinline__ void async_copy_b128(void* lds_dst, const void* gsrc) {
  // LDS aperture: low 32 bits of the generic address are the LDS byte address
  unsigned loff = (unsigned)(size_t)lds_dst;
  unsigned long long ga = (unsigned long long)(size_t)gsrc;
  asm volatile("global_load_async_to_lds_b128 %0, %1, off"
               :: "v"(loff), "v"(ga)
               : "memory");
}
__device__ __forceinline__ void wait_async0() {
  asm volatile("s_wait_asynccnt 0x0" ::: "memory");
}

// ---------- problem constants ----------
#define N_EMBD 1024
#define N_HEAD 16
#define HEADD  64
#define SEQ_B  4
#define SEQ_T  2048
#define C3     (3 * N_EMBD)

// ---------- GEMM: C[M,N] = A[M,K] * W[K,N] + bias, bf16 WMMA ----------
#define TM   128
#define TN   128
#define TK   32
#define LSTR 40   // padded LDS k-stride (bf16 elems); 80B = 16B-aligned rows

template <bool A_BF16, bool OUT_F32>
__global__ __launch_bounds__(256) void gemm_kernel(
    const void* __restrict__ A_, const float* __restrict__ W,
    const float* __restrict__ bias, void* __restrict__ Out_,
    int M, int N, int K) {
  __shared__ unsigned short As[TM * LSTR];   // A tile [m][k] bf16
  __shared__ unsigned short Bts[TN * LSTR];  // B tile transposed [n][k] bf16

  const int tid  = threadIdx.x;
  const int lane = tid & 31;
  const int wv   = tid >> 5;   // 8 waves
  const int wm   = wv & 3;     // 4 M-groups of 32 rows
  const int wn   = wv >> 2;    // 2 N-groups of 64 cols
  const int lr   = lane & 15;
  const int hi   = lane >> 4;  // lane half
  const int kb   = hi * 8;     // A-fragment K base

  const int m0 = blockIdx.y * TM;
  const int n0 = blockIdx.x * TN;

  v8f zero = {};
  v8f acc[2][4];
  for (int i = 0; i < 2; ++i)
    for (int j = 0; j < 4; ++j) acc[i][j] = zero;

  const int ar = tid >> 1;          // A staging: row 0..127
  const int ak = (tid & 1) * 16;    // k sub-block 0/16
  const int br = tid >> 3;          // B staging: k-row 0..31
  const int bn = (tid & 7) * 16;    // n base 0..112

  for (int k0 = 0; k0 < K; k0 += TK) {
    // prefetch next K-step's f32 W rows into cache (global_prefetch_b8)
    if (k0 + TK < K) {
      if (!A_BF16)
        __builtin_prefetch((const float*)A_ +
                               (size_t)(m0 + ar) * K + (k0 + TK) + ak, 0, 1);
      __builtin_prefetch(W + (size_t)(k0 + TK + br) * N + n0 + bn, 0, 1);
    }
    __syncthreads();
    // ---- stage A ----
    if (A_BF16) {
      // bf16 A: async DMA straight into LDS (no VGPR round trip)
      const unsigned short* g =
          (const unsigned short*)A_ + (size_t)(m0 + ar) * K + k0 + ak;
      async_copy_b128(As + ar * LSTR + ak, g);
      async_copy_b128(As + ar * LSTR + ak + 8, g + 8);
    } else {
      const float* g = (const float*)A_ + (size_t)(m0 + ar) * K + k0 + ak;
      unsigned short* s = As + ar * LSTR + ak;
      for (int i = 0; i < 4; ++i) {
        f32x4 v = *(const f32x4*)(g + i * 4);
        u16x4 p;
        p.x = f2bf_raw(v.x); p.y = f2bf_raw(v.y);
        p.z = f2bf_raw(v.z); p.w = f2bf_raw(v.w);
        *(u16x4*)(s + i * 4) = p;
      }
    }
    // ---- stage B transposed: Bts[n][k] (f32 -> bf16 convert) ----
    {
      const float* g = W + (size_t)(k0 + br) * N + n0 + bn;
      for (int i = 0; i < 16; i += 4) {
        f32x4 v = *(const f32x4*)(g + i);
        Bts[(bn + i + 0) * LSTR + br] = f2bf_raw(v.x);
        Bts[(bn + i + 1) * LSTR + br] = f2bf_raw(v.y);
        Bts[(bn + i + 2) * LSTR + br] = f2bf_raw(v.z);
        Bts[(bn + i + 3) * LSTR + br] = f2bf_raw(v.w);
      }
    }
    if (A_BF16) wait_async0();
    __syncthreads();

    // ---- fragments + 8 WMMAs per wave ----
    v16bf af[2], bfr[4];
    for (int mt = 0; mt < 2; ++mt) {
      const unsigned short* p = As + (wm * 32 + mt * 16 + lr) * LSTR;
      af[mt] = ld_frag_pair(p + kb, p + kb + 16);  // K {kb..kb+7, kb+16..kb+23}
    }
    for (int nt = 0; nt < 4; ++nt) {
      const unsigned short* p =
          Bts + (wn * 64 + nt * 16 + lr) * LSTR + kb;  // K {kb..kb+7, kb+16..kb+23}
      bfr[nt] = ld_frag_pair(p, p + 16);
    }
    for (int mt = 0; mt < 2; ++mt)
      for (int nt = 0; nt < 4; ++nt)
        acc[mt][nt] = wmma_bf(af[mt], bfr[nt], acc[mt][nt]);
  }

  // ---- epilogue: + bias, store ----
  for (int mt = 0; mt < 2; ++mt)
    for (int nt = 0; nt < 4; ++nt) {
      const int ng = n0 + wn * 64 + nt * 16 + lr;
      const float bv = bias[ng];
      const int mg = m0 + wm * 32 + mt * 16 + hi * 8;
      for (int j = 0; j < 8; ++j) {
        const float v = acc[mt][nt][j] + bv;
        if (OUT_F32)
          ((float*)Out_)[(size_t)(mg + j) * N + ng] = v;
        else
          ((unsigned short*)Out_)[(size_t)(mg + j) * N + ng] = f2bf_raw(v);
      }
    }
}

// ---------- flash attention (bf16 WMMA, f32 online softmax) ----------
#define KTILE 64
#define KSTR  72   // 144B rows, 16B-aligned
#define SSTR  65   // f32 score row stride (conflict-skewed)

__global__ __launch_bounds__(256) void attn_kernel(
    const unsigned short* __restrict__ qkv,   // [B*T, 3C] bf16
    unsigned short* __restrict__ aout) {      // [B*T, C] bf16
  __shared__ unsigned short Ks[KTILE * KSTR];   // K tile [key][d]
  __shared__ unsigned short Vts[HEADD * KSTR];  // V tile transposed [d][key]
  __shared__ float Ssm[8 * 16 * SSTR];          // per-wave score/P buffer
  __shared__ float corr_sm[8 * 16];
  __shared__ float lsum_sm[8 * 16];

  const int tid = threadIdx.x, lane = tid & 31, wv = tid >> 5;
  const int lr = lane & 15, hi = lane >> 4, kb = hi * 8;
  const int qt = blockIdx.x, h = blockIdx.y, b = blockIdx.z;
  const int q0 = qt * 128;
  const int tokbase = b * SEQ_T;

  // Q fragments: 16 rows per wave, two K=32 chunks over d=64, straight from global
  v16bf qf[2];
  {
    const unsigned short* qp =
        qkv + (size_t)(tokbase + q0 + wv * 16 + lr) * C3 + h * HEADD;
    for (int c = 0; c < 2; ++c)
      qf[c] = ld_frag_pair(qp + c * 32 + kb, qp + c * 32 + kb + 16);
  }

  v8f zero = {};
  v8f oacc[4];
  for (int t = 0; t < 4; ++t) oacc[t] = zero;
  float m_r = -1e30f, l_r = 0.0f;   // row max / row sum (lanes 0-15 own a row)
  const float scale = 0.125f;       // 1/sqrt(64)
  float* Sw = Ssm + wv * (16 * SSTR);

  const int ntiles = (q0 + 128) / KTILE;  // causal upper bound for this block

  for (int kt = 0; kt < ntiles; ++kt) {
    // prefetch next V tile rows (V goes through VGPRs for the transpose)
    if (kt + 1 < ntiles) {
      const unsigned short* nvp =
          qkv + (size_t)(tokbase + (kt + 1) * KTILE + (tid >> 2)) * C3 +
          2 * N_EMBD + h * HEADD + (tid & 3) * 16;
      __builtin_prefetch(nvp, 0, 1);
    }
    __syncthreads();
    // ---- stage K [key][d] (async DMA) and V transposed [d][key] ----
    {
      const int key = tid >> 2;
      const int d0 = (tid & 3) * 16;
      const unsigned short* kp =
          qkv + (size_t)(tokbase + kt * KTILE + key) * C3 + N_EMBD + h * HEADD + d0;
      async_copy_b128(Ks + key * KSTR + d0, kp);
      async_copy_b128(Ks + key * KSTR + d0 + 8, kp + 8);
      const unsigned short* vp = kp + N_EMBD;  // V is next C block
      FragBF tv;
      tv.h[0] = *(const u16x8*)vp;
      tv.h[1] = *(const u16x8*)(vp + 8);
      for (int i = 0; i < 16; ++i) Vts[(d0 + i) * KSTR + key] = tv.e[i];
    }
    wait_async0();
    __syncthreads();

    // ---- S = Q * K^T (4 key subtiles, K-depth 64 = 2 WMMA) ----
    for (int nt = 0; nt < 4; ++nt) {
      const unsigned short* p = Ks + (nt * 16 + lr) * KSTR + kb;
      v16bf kf0 = ld_frag_pair(p, p + 16);
      v16bf kf1 = ld_frag_pair(p + 32, p + 32 + 16);
      v8f s = wmma_bf(qf[0], kf0, zero);
      s = wmma_bf(qf[1], kf1, s);
      for (int j = 0; j < 8; ++j)
        Sw[(hi * 8 + j) * SSTR + nt * 16 + lr] = s[j];
    }
    __syncthreads();

    // ---- online softmax row scan (lane r owns q-row r; causal mask here) ----
    if (lane < 16) {
      float* rp = Sw + lane * SSTR;
      const int myq = q0 + wv * 16 + lane;
      int nvalid = myq - kt * KTILE + 1;       // keys <= myq in this tile
      if (nvalid > KTILE) nvalid = KTILE;
      float tmax = -1e30f;
      for (int j = 0; j < nvalid; ++j) {
        const float v = rp[j] * scale;
        rp[j] = v;
        tmax = v > tmax ? v : tmax;
      }
      const float m_new = m_r > tmax ? m_r : tmax;
      const float corr = __expf(m_r - m_new);
      float ls = 0.0f;
      int j = 0;
      for (; j < nvalid; ++j) {
        const float p = __expf(rp[j] - m_new);
        rp[j] = p;            // P overwrites S in place
        ls += p;
      }
      for (; j < KTILE; ++j) rp[j] = 0.0f;  // masked keys
      l_r = l_r * corr + ls;
      m_r = m_new;
      corr_sm[wv * 16 + lane] = corr;
    }
    __syncthreads();

    // ---- rescale O accumulators by per-row correction ----
    {
      const float* cp = corr_sm + wv * 16 + hi * 8;
      f32x4 c0 = *(const f32x4*)cp;
      f32x4 c1 = *(const f32x4*)(cp + 4);
      const float cf[8] = {c0.x, c0.y, c0.z, c0.w, c1.x, c1.y, c1.z, c1.w};
      for (int t = 0; t < 4; ++t)
        for (int j = 0; j < 8; ++j) oacc[t][j] *= cf[j];
    }

    // ---- P fragments (A layout) from LDS, then O += P * V ----
    v16bf pf[2];
    for (int c = 0; c < 2; ++c) {
      FragBF f;
      const float* pr = Sw + lr * SSTR + c * 32 + kb;
      for (int e = 0; e < 8; ++e) f.e[e]     = f2bf_raw(pr[e]);
      for (int e = 0; e < 8; ++e) f.e[8 + e] = f2bf_raw(pr[16 + e]);
      pf[c] = f.v;
    }
    for (int t = 0; t < 4; ++t) {
      const unsigned short* p = Vts + (t * 16 + lr) * KSTR + kb;
      v16bf vf0 = ld_frag_pair(p, p + 16);
      v16bf vf1 = ld_frag_pair(p + 32, p + 32 + 16);
      oacc[t] = wmma_bf(pf[0], vf0, oacc[t]);
      oacc[t] = wmma_bf(pf[1], vf1, oacc[t]);
    }
  }

  // ---- epilogue: O / l, bf16 store to [B*T, C] with col = h*64 + d ----
  if (lane < 16) lsum_sm[wv * 16 + lane] = l_r;
  __syncthreads();
  {
    const float* lp = lsum_sm + wv * 16 + hi * 8;
    f32x4 l0 = *(const f32x4*)lp;
    f32x4 l1 = *(const f32x4*)(lp + 4);
    const float lf[8] = {l0.x, l0.y, l0.z, l0.w, l1.x, l1.y, l1.z, l1.w};
    for (int t = 0; t < 4; ++t)
      for (int j = 0; j < 8; ++j) {
        const int tok = tokbase + q0 + wv * 16 + hi * 8 + j;
        const int col = h * HEADD + t * 16 + lr;
        aout[(size_t)tok * N_EMBD + col] = f2bf_raw(oacc[t][j] / lf[j]);
      }
  }
}

// ---------- launcher ----------
extern "C" void kernel_launch(void* const* d_in, const int* in_sizes, int n_in,
                              void* d_out, int out_size, void* d_ws,
                              size_t ws_size, hipStream_t stream) {
  (void)in_sizes; (void)n_in; (void)out_size; (void)ws_size;
  const float* x      = (const float*)d_in[0];  // [B,T,C]
  const float* w_attn = (const float*)d_in[1];  // [C,3C]
  const float* b_attn = (const float*)d_in[2];  // [3C]
  const float* w_proj = (const float*)d_in[3];  // [C,C]
  const float* b_proj = (const float*)d_in[4];  // [C]
  float* out = (float*)d_out;                   // [B,T,C] f32

  // workspace: qkv bf16 [8192,3072] (48MB) + attn_out bf16 [8192,1024] (16MB)
  unsigned short* qkv  = (unsigned short*)d_ws;
  unsigned short* aout = qkv + (size_t)SEQ_B * SEQ_T * C3;

  const int M = SEQ_B * SEQ_T;  // 8192
  dim3 blk(256);

  // 1) QKV projection: x @ w_attn + b_attn -> bf16
  gemm_kernel<false, false><<<dim3(C3 / TN, M / TM), blk, 0, stream>>>(
      (const void*)x, w_attn, b_attn, (void*)qkv, M, C3, N_EMBD);

  // 2) causal flash attention per (b,h,128-row q tile)
  attn_kernel<<<dim3(SEQ_T / 128, N_HEAD, SEQ_B), blk, 0, stream>>>(qkv, aout);

  // 3) output projection: aout @ w_proj + b_proj -> f32
  gemm_kernel<true, true><<<dim3(N_EMBD / TN, M / TM), blk, 0, stream>>>(
      (const void*)aout, w_proj, b_proj, (void*)out, M, N_EMBD, N_EMBD);
}
